// SparseMoELayer_80857054314539
// MI455X (gfx1250) — matrix operand order repaired and checked by reading, not compiled, not run on hardware
//
#include <hip/hip_runtime.h>
#include <hip/hip_bf16.h>

// ---------------------------------------------------------------------------
// Sparse MoE (top-2 of 8 experts), H=1024, I=4096, T=8192 tokens.
// MI455X strategy:
//   * bf16 WMMA (v_wmma_f32_16x16x32_bf16), f32 accumulate
//   * weights fp32->bf16 + transpose to K-major once; 134 MB bf16 lives in L2
//   * gather tokens per expert -> grouped GEMMs on routed rows only
//   * fused GEMM1+GELU+GEMM2 per 32-row M-tile; intermediate chunked 8x512 in LDS
//   * M=32 register blocking: each B fragment feeds 2 WMMAs (halves L2 traffic)
// ---------------------------------------------------------------------------

#define HD    1024
#define ID    4096
#define NE    8
#define NT    8192          // tokens (2*4096)
#define GTOT  (2*NT)        // gathered rows total (top-2)
#define MT    32            // M-tile rows per workgroup
#define CHUNK 512
#define LDA   (HD + 8)      // LDS row stride (halfs), bank-rotating pad
#define LDH   (CHUNK + 8)

typedef __attribute__((ext_vector_type(16))) __bf16 v16bf;
typedef __attribute__((ext_vector_type(8)))  __bf16 v8bf;
typedef __attribute__((ext_vector_type(8)))  float  v8f;

union AFrag { v16bf v; v8bf h[2]; };

static __device__ __forceinline__ unsigned short f2bf(float f) {
    unsigned int u = __float_as_uint(f);
    u = u + 0x7FFFu + ((u >> 16) & 1u);     // round-to-nearest-even
    return (unsigned short)(u >> 16);
}

// ---------------------------------------------------------------- zero kernels
__global__ void zero_f32_kernel(float* __restrict__ p, size_t n) {
    size_t i = (size_t)blockIdx.x * blockDim.x + threadIdx.x;
    if (i < n) p[i] = 0.0f;
}

__global__ void zero_meta_kernel(int* __restrict__ meta, float* __restrict__ imp) {
    int t = threadIdx.x;
    if (t < 64) meta[t] = 0;
    if (t < NE) imp[t] = 0.0f;
}

// ------------------------------------------------------------ fp32 -> bf16 (x)
__global__ void cvt_bf16_kernel(const float* __restrict__ src,
                                unsigned short* __restrict__ dst, size_t n4) {
    size_t i = (size_t)blockIdx.x * blockDim.x + threadIdx.x;
    if (i >= n4) return;
    float4 v = ((const float4*)src)[i];
    ushort4 o;
    o.x = f2bf(v.x); o.y = f2bf(v.y); o.z = f2bf(v.z); o.w = f2bf(v.w);
    ((ushort4*)dst)[i] = o;
}

// -------------------------------------- fp32 [E][R][C] -> bf16 transposed [E][C][R]
__global__ __launch_bounds__(256)
void transpose_cvt_kernel(const float* __restrict__ src,
                          unsigned short* __restrict__ dst, int R, int C) {
    __shared__ float tile[32][33];
    int e = blockIdx.z;
    const float* s = src + (size_t)e * R * C;
    unsigned short* d = dst + (size_t)e * R * C;
    int x  = blockIdx.x * 32 + threadIdx.x;       // src col
    int y0 = blockIdx.y * 32 + threadIdx.y;       // src row base
#pragma unroll
    for (int i = 0; i < 32; i += 8)
        tile[threadIdx.y + i][threadIdx.x] = s[(size_t)(y0 + i) * C + x];
    __syncthreads();
    int xo  = blockIdx.y * 32 + threadIdx.x;      // dst col = src row
    int yo0 = blockIdx.x * 32 + threadIdx.y;      // dst row = src col
#pragma unroll
    for (int i = 0; i < 32; i += 8)
        d[(size_t)(yo0 + i) * R + xo] = f2bf(tile[threadIdx.x][threadIdx.y + i]);
}

// ---------------------------------------------------------------- router
// meta[0..7]=counts  meta[8..15]=cursor  meta[16..24]=offsets  meta[32..40]=tile offs
__global__ __launch_bounds__(256)
void router_kernel(const float* __restrict__ x, const float* __restrict__ rw,
                   const float* __restrict__ rb, int2* __restrict__ topidx,
                   float2* __restrict__ topw, int* __restrict__ meta,
                   float* __restrict__ importance) {
    __shared__ float s_imp[NE];
    __shared__ int   s_cnt[NE];
    int tid = threadIdx.x;
    if (tid < NE) { s_imp[tid] = 0.0f; s_cnt[tid] = 0; }
    __syncthreads();

    int t = blockIdx.x * 256 + tid;
    float logit[NE];
#pragma unroll
    for (int e = 0; e < NE; e++) logit[e] = rb[e];
    const float* xr = x + (size_t)t * HD;
    for (int i = 0; i < HD; i++) {
        float xi = xr[i];
#pragma unroll
        for (int e = 0; e < NE; e++) logit[e] += xi * rw[i * NE + e];
    }
    // top-2 (first occurrence on ties, matching lax.top_k order)
    int i0 = 0; float v0 = logit[0];
#pragma unroll
    for (int e = 1; e < NE; e++) if (logit[e] > v0) { v0 = logit[e]; i0 = e; }
    int i1 = (i0 == 0) ? 1 : 0; float v1 = logit[i1];
#pragma unroll
    for (int e = 0; e < NE; e++)
        if (e != i0 && logit[e] > v1) { v1 = logit[e]; i1 = e; }

    float w0 = 1.0f / (1.0f + __expf(v1 - v0));   // softmax over [v0, v1]
    float w1 = 1.0f - w0;
    topidx[t] = make_int2(i0, i1);
    topw[t]   = make_float2(w0, w1);

    // full softmax for importance (aux loss)
    float s = 0.0f, p[NE];
#pragma unroll
    for (int e = 0; e < NE; e++) { p[e] = __expf(logit[e] - v0); s += p[e]; }
    float inv = 1.0f / s;
#pragma unroll
    for (int e = 0; e < NE; e++) atomicAdd(&s_imp[e], p[e] * inv);
    atomicAdd(&s_cnt[i0], 1);
    atomicAdd(&s_cnt[i1], 1);
    __syncthreads();
    if (tid < NE) {
        atomicAdd(&importance[tid], s_imp[tid]);
        atomicAdd(&meta[tid], s_cnt[tid]);
    }
}

// ------------------------------------------------- prefix sums + aux loss
__global__ void finalize_kernel(int* __restrict__ meta,
                                const float* __restrict__ importance,
                                float* __restrict__ aux_out) {
    if (threadIdx.x != 0) return;
    meta[16] = 0;
    meta[32] = 0;
    for (int e = 0; e < NE; e++) {
        meta[16 + e + 1] = meta[16 + e] + meta[e];
        meta[32 + e + 1] = meta[32 + e] + (meta[e] + MT - 1) / MT;
    }
    float simp = 0.0f, sload = 0.0f;
    for (int e = 0; e < NE; e++) { simp += importance[e]; sload += (float)meta[e]; }
    float aux = 0.0f;
    for (int e = 0; e < NE; e++)
        aux += (importance[e] / simp) * ((float)meta[e] / sload);
    aux_out[0] = aux * (float)NE;
}

// ---------------------------------------------------------------- scatter
__global__ __launch_bounds__(256)
void scatter_kernel(const int2* __restrict__ topidx, const float2* __restrict__ topw,
                    int* __restrict__ meta, int* __restrict__ perm,
                    float* __restrict__ gate) {
    int t = blockIdx.x * 256 + threadIdx.x;
    int2 ti = topidx[t];
    float2 tw = topw[t];
    int* cursor = meta + 8;
    const int* offsets = meta + 16;
    int p0 = atomicAdd(&cursor[ti.x], 1);
    perm[offsets[ti.x] + p0] = t;
    gate[offsets[ti.x] + p0] = tw.x;
    int p1 = atomicAdd(&cursor[ti.y], 1);
    perm[offsets[ti.y] + p1] = t;
    gate[offsets[ti.y] + p1] = tw.y;
}

// ------------------------------------------------ fused expert FFN (WMMA bf16)
// One block = one 32-row gathered M-tile of one expert. 8 waves.
// Each wave multiplies one B fragment against TWO A fragments (rows 0-15/16-31),
// halving weight-stream traffic per WMMA.
__global__ __launch_bounds__(256)
void moe_ffn_kernel(const unsigned short* __restrict__ xb,
                    const unsigned short* __restrict__ w1t,   // [E][ID][HD] bf16
                    const unsigned short* __restrict__ w2t,   // [E][HD][ID] bf16
                    const float* __restrict__ b1,             // [E][ID]
                    const float* __restrict__ b2,             // [E][HD]
                    const int* __restrict__ perm,
                    const float* __restrict__ gate,
                    const int* __restrict__ meta,
                    float* __restrict__ out) {
    const int* tileoff = meta + 32;
    int bid = blockIdx.x;
    if (bid >= tileoff[8]) return;

    int e = 0;
#pragma unroll
    for (int k = 1; k < NE; k++) if (bid >= tileoff[k]) e = k;
    const int tile    = bid - tileoff[e];
    const int cnt     = meta[e];
    const int segbase = meta[16 + e];
    const int rowbase = tile * MT;

    __shared__ unsigned short As[MT * LDA];   // 66 KB (32 x 1024 bf16, padded)
    __shared__ unsigned short Hs[MT * LDH];   // 33 KB (32 x 512 bf16, padded)

    const int tid = threadIdx.x;

    // ---- stage gathered A rows (bf16 x) into LDS: 32 rows x 128 x 16B chunks
    for (int c = tid; c < MT * 128; c += 256) {
        int r  = c >> 7;
        int ch = c & 127;
        int gr = rowbase + r;
        int tok = (gr < cnt) ? perm[segbase + gr] : 0;
        uint4 v = *(const uint4*)(xb + (size_t)tok * HD + ch * 8);
        *(uint4*)(&As[r * LDA + ch * 8]) = v;
    }

    const int lane = tid & 31;
    const int wave = tid >> 5;                 // 0..7
    const int nlo  = lane & 15;                // N (and A row-within-slab) index
    const int khi  = (lane < 16) ? 0 : 16;     // B-fragment K offset
    const int asel = (lane < 16) ? 0 : 8;      // D row offset

    // persistent GEMM2 accumulators: wave owns out cols [wave*128, wave*128+128)
    // for both 16-row M-slabs
    v8f accO[2][8];
#pragma unroll
    for (int nt = 0; nt < 8; nt++) {
        float bv = b2[e * HD + wave * 128 + nt * 16 + nlo];
#pragma unroll
        for (int v = 0; v < 8; v++) { accO[0][nt][v] = bv; accO[1][nt][v] = bv; }
    }

    const unsigned short* w1base = w1t + (size_t)e * ID * HD;
    const unsigned short* w2base = w2t + (size_t)e * HD * ID;

    __syncthreads();   // A tile staged

    for (int ch = 0; ch < ID / CHUNK; ch++) {
        const int cbase = ch * CHUNK;
        __syncthreads();   // previous chunk's GEMM2 done reading Hs

        // ---- GEMM1: wave computes h cols [wave*64, wave*64+64), 2 passes of 2 tiles
#pragma unroll
        for (int g = 0; g < 2; g++) {
            v8f acc1[2][2];
#pragma unroll
            for (int j = 0; j < 2; j++) {
                float bv = b1[e * ID + cbase + wave * 64 + (g * 2 + j) * 16 + nlo];
#pragma unroll
                for (int v = 0; v < 8; v++) { acc1[0][j][v] = bv; acc1[1][j][v] = bv; }
            }
            for (int k0 = 0; k0 < HD; k0 += 32) {
                AFrag a0, a1;
                a0.h[0] = *(const v8bf*)(&As[nlo * LDA + k0 + khi]);
                a0.h[1] = *(const v8bf*)(&As[nlo * LDA + k0 + khi + 8]);
                a1.h[0] = *(const v8bf*)(&As[(16 + nlo) * LDA + k0 + khi]);
                a1.h[1] = *(const v8bf*)(&As[(16 + nlo) * LDA + k0 + khi + 8]);
#pragma unroll
                for (int j = 0; j < 2; j++) {
                    int ncol = cbase + wave * 64 + (g * 2 + j) * 16 + nlo;
                    v16bf b = *(const v16bf*)(w1base + (size_t)ncol * HD + k0 + khi);
                    acc1[0][j] = __builtin_amdgcn_wmma_f32_16x16x32_bf16(
                        false, a0.v, false, b, (short)0, acc1[0][j], false, false);
                    acc1[1][j] = __builtin_amdgcn_wmma_f32_16x16x32_bf16(
                        false, a1.v, false, b, (short)0, acc1[1][j], false, false);
                }
            }
            // exact GELU -> bf16 -> LDS h chunk
#pragma unroll
            for (int ms = 0; ms < 2; ms++) {
#pragma unroll
                for (int j = 0; j < 2; j++) {
#pragma unroll
                    for (int v = 0; v < 8; v++) {
                        float xg = acc1[ms][j][v];
                        float gl = 0.5f * xg * (1.0f + erff(xg * 0.70710678118f));
                        int m = ms * 16 + v + asel;
                        Hs[m * LDH + wave * 64 + (g * 2 + j) * 16 + nlo] = f2bf(gl);
                    }
                }
            }
        }
        __syncthreads();   // Hs ready

        // ---- GEMM2: accumulate h_chunk @ w2t into persistent accO
        for (int k0 = 0; k0 < CHUNK; k0 += 32) {
            AFrag a0, a1;
            a0.h[0] = *(const v8bf*)(&Hs[nlo * LDH + k0 + khi]);
            a0.h[1] = *(const v8bf*)(&Hs[nlo * LDH + k0 + khi + 8]);
            a1.h[0] = *(const v8bf*)(&Hs[(16 + nlo) * LDH + k0 + khi]);
            a1.h[1] = *(const v8bf*)(&Hs[(16 + nlo) * LDH + k0 + khi + 8]);
            int kg = cbase + k0 + khi;
#pragma unroll
            for (int nt = 0; nt < 8; nt++) {
                int ncol = wave * 128 + nt * 16 + nlo;
                v16bf b = *(const v16bf*)(w2base + (size_t)ncol * ID + kg);
                accO[0][nt] = __builtin_amdgcn_wmma_f32_16x16x32_bf16(
                    false, a0.v, false, b, (short)0, accO[0][nt], false, false);
                accO[1][nt] = __builtin_amdgcn_wmma_f32_16x16x32_bf16(
                    false, a1.v, false, b, (short)0, accO[1][nt], false, false);
            }
        }
    }

    // ---- gate-scale and accumulate into output (each token hit by 2 experts)
#pragma unroll
    for (int ms = 0; ms < 2; ms++) {
#pragma unroll
        for (int v = 0; v < 8; v++) {
            int m  = ms * 16 + v + asel;
            int gr = rowbase + m;
            if (gr < cnt) {
                int   tok = perm[segbase + gr];
                float gw  = gate[segbase + gr];
#pragma unroll
                for (int nt = 0; nt < 8; nt++) {
                    int col = wave * 128 + nt * 16 + nlo;
                    atomicAdd(&out[(size_t)tok * HD + col], gw * accO[ms][nt][v]);
                }
            }
        }
    }
}

// ---------------------------------------------------------------------------
extern "C" void kernel_launch(void* const* d_in, const int* in_sizes, int n_in,
                              void* d_out, int out_size, void* d_ws, size_t ws_size,
                              hipStream_t stream) {
    const float* x  = (const float*)d_in[0];   // [2,4096,1024]
    const float* rw = (const float*)d_in[1];   // [1024,8]
    const float* rb = (const float*)d_in[2];   // [8]
    const float* w1 = (const float*)d_in[3];   // [8,1024,4096]
    const float* b1 = (const float*)d_in[4];   // [8,4096]
    const float* w2 = (const float*)d_in[5];   // [8,4096,1024]
    const float* b2 = (const float*)d_in[6];   // [8,1024]
    float* out = (float*)d_out;                // 8192*1024 output + 1 aux

    // workspace layout (~151.3 MB)
    unsigned char* ws = (unsigned char*)d_ws;
    size_t o = 0;
    unsigned short* xb  = (unsigned short*)(ws + o); o += (size_t)NT * HD * 2;        // 16.8 MB
    unsigned short* w1t = (unsigned short*)(ws + o); o += (size_t)NE * HD * ID * 2;   // 67.1 MB
    unsigned short* w2t = (unsigned short*)(ws + o); o += (size_t)NE * HD * ID * 2;   // 67.1 MB
    int2*   topidx = (int2*)  (ws + o); o += (size_t)NT * 8;
    float2* topw   = (float2*)(ws + o); o += (size_t)NT * 8;
    int*    perm   = (int*)   (ws + o); o += (size_t)(GTOT + 64) * 4;
    float*  gate   = (float*) (ws + o); o += (size_t)(GTOT + 64) * 4;
    int*    meta   = (int*)   (ws + o); o += 256;
    float*  imp    = (float*) (ws + o); o += 256;
    (void)ws_size; (void)in_sizes; (void)n_in;

    const size_t outN = (size_t)out_size;

    // 1) zero output (atomic-accumulated) and meta/importance
    zero_f32_kernel<<<(unsigned)((outN + 255) / 256), 256, 0, stream>>>(out, outN);
    zero_meta_kernel<<<1, 64, 0, stream>>>(meta, imp);

    // 2) x -> bf16
    cvt_bf16_kernel<<<(unsigned)(((size_t)NT * HD / 4 + 255) / 256), 256, 0, stream>>>(
        x, xb, (size_t)NT * HD / 4);

    // 3) weights -> bf16, transposed to K-major (fits in 192 MB L2)
    transpose_cvt_kernel<<<dim3(ID / 32, HD / 32, NE), dim3(32, 8), 0, stream>>>(
        w1, w1t, HD, ID);    // [E][HD][ID] -> [E][ID][HD]
    transpose_cvt_kernel<<<dim3(HD / 32, ID / 32, NE), dim3(32, 8), 0, stream>>>(
        w2, w2t, ID, HD);    // [E][ID][HD] -> [E][HD][ID]

    // 4) router: logits, top-2, gates, importance, expert counts
    router_kernel<<<NT / 256, 256, 0, stream>>>(x, rw, rb, topidx, topw, meta, imp);

    // 5) prefix sums + aux loss (last output element)
    finalize_kernel<<<1, 32, 0, stream>>>(meta, imp, out + (outN - 1));

    // 6) compact token lists per expert
    scatter_kernel<<<NT / 256, 256, 0, stream>>>(topidx, topw, meta, perm, gate);

    // 7) fused grouped-GEMM expert FFN: worst case GTOT/MT + NE tiles
    moe_ffn_kernel<<<GTOT / MT + NE, 256, 0, stream>>>(
        xb, w1t, w2t, b1, b2, perm, gate, meta, out);
}
